// Model_75265006895519
// MI455X (gfx1250) — matrix operand order repaired and checked by reading, not compiled, not run on hardware
//
#include <hip/hip_runtime.h>
#include <hip/hip_bf16.h>

#define Bdim   8
#define Tdim   512
#define Ddim   256
#define DIdim  512
#define KCONV  4
#define Mdim   2
#define Hdim   4
#define HDdim  64
#define Ldim   2
#define VOCABd 272
#define NT     (Bdim*Tdim)   /* 4096 tokens */

typedef __bf16 bf16_t;
typedef __attribute__((ext_vector_type(16))) bf16_t v16bf;
typedef __attribute__((ext_vector_type(8)))  float  v8f;

__device__ __forceinline__ unsigned short f2bf(float f) {
  unsigned u = __float_as_uint(f);
  u += 0x7FFFu + ((u >> 16) & 1u);           // round-to-nearest-even
  return (unsigned short)(u >> 16);
}
__device__ __forceinline__ float sigmoidf_(float x) { return 1.0f / (1.0f + expf(-x)); }

// ---------------------------------------------------------------- embedding
__global__ __launch_bounds__(256) void embed_kernel(const int* __restrict__ ids,
                                                    const float* __restrict__ emb,
                                                    float* __restrict__ x) {
  int i = blockIdx.x * blockDim.x + threadIdx.x;
  if (i >= NT * Ddim) return;
  int tok = i >> 8, d = i & 255;
  x[i] = emb[(size_t)ids[tok] * Ddim + d];
}

// ------------------------------------------------------- f32 -> bf16 (plain)
__global__ __launch_bounds__(256) void cvt_bf16_kernel(const float* __restrict__ in,
                                                       unsigned short* __restrict__ out, int n) {
  int i = blockIdx.x * blockDim.x + threadIdx.x;
  if (i < n) out[i] = f2bf(in[i]);
}

// ---------------------------------------- f32 KxN -> bf16 NxK (transposed B)
__global__ __launch_bounds__(256) void cvtT_bf16_kernel(const float* __restrict__ in,
                                                        unsigned short* __restrict__ out,
                                                        int Kd, int N) {
  int i = blockIdx.x * blockDim.x + threadIdx.x;
  if (i >= Kd * N) return;
  int n = i / Kd, k = i - n * Kd;
  out[i] = f2bf(in[(size_t)k * N + n]);
}

// ------------------------------------------------------------------ rmsnorm
__global__ __launch_bounds__(256) void rmsnorm_kernel(const float* __restrict__ x,
                                                      const float* __restrict__ w,
                                                      float* __restrict__ y,
                                                      unsigned short* __restrict__ ybf) {
  int row = blockIdx.x, tid = threadIdx.x;
  __shared__ float red[256];
  float v = x[(size_t)row * Ddim + tid];
  red[tid] = v * v;
  __syncthreads();
  for (int s = 128; s > 0; s >>= 1) {
    if (tid < s) red[tid] += red[tid + s];
    __syncthreads();
  }
  float inv = rsqrtf(red[0] * (1.0f / Ddim) + 1e-6f);
  float o = v * inv * w[tid];
  y[(size_t)row * Ddim + tid] = o;
  ybf[(size_t)row * Ddim + tid] = f2bf(o);
}

// ---------------- fragment loaders (CDNA5 ISA 7.12.2 wave32 layouts) -------
__device__ __forceinline__ v16bf load_a_frag(const unsigned short* Ar, int k0, int half) {
  union { unsigned u[8]; v16bf v; } ua;
#pragma unroll
  for (int vv = 0; vv < 8; ++vv) {
    // A 16x32 bf16: lane halves split K; VGPR vv holds a packed pair
    int ka = k0 + ((vv < 4) ? 0 : 16) + ((vv & 3) << 1) + (half << 3);
    ua.u[vv] = *(const unsigned*)(Ar + ka);
  }
  return ua.v;
}
__device__ __forceinline__ v16bf load_b_frag(const unsigned short* Br, int k0, int half) {
  union { unsigned u[8]; v16bf v; } ub;
#pragma unroll
  for (int vv = 0; vv < 8; ++vv) {
    // B 32x16 bf16 from transposed weights: lane halves split K like A
    int kb = k0 + ((vv < 4) ? 0 : 16) + ((vv & 3) << 1) + (half << 3);
    ub.u[vv] = *(const unsigned*)(Br + kb);
  }
  return ub.v;
}

// ---------- WMMA bf16 GEMM, 16x64 tile/wave (requires (N/16) % 4 == 0) -----
// Straight-line: no guards anywhere near v_wmma, EXEC stays all-ones.
__global__ __launch_bounds__(256) void gemm_bf16_wmma4(const unsigned short* __restrict__ A,
                                                       const unsigned short* __restrict__ BT,
                                                       const float* __restrict__ add,
                                                       float* __restrict__ C,
                                                       int Mr, int N, int Kd) {
  const int wave = blockIdx.x * (blockDim.x >> 5) + (threadIdx.x >> 5);
  const int tilesN4 = N >> 6;
  const int tiles = (Mr >> 4) * tilesN4;
  if (wave >= tiles) return;                 // whole waves drop out only
  const int tm  = wave / tilesN4;
  const int tn4 = wave - tm * tilesN4;
  const int lane = threadIdx.x & 31;
  const int half = lane >> 4;
  const int l16  = lane & 15;
  const unsigned short* Ar = A  + (size_t)(tm * 16 + l16) * Kd;
  const unsigned short* Bb = BT + (size_t)(tn4 * 64 + l16) * Kd;   // sub-tiles at +j*16*Kd
  v8f acc0 = {}, acc1 = {}, acc2 = {}, acc3 = {};
  for (int k0 = 0; k0 < Kd; k0 += 32) {
    v16bf a = load_a_frag(Ar, k0, half);
    if (k0 + 64 < Kd) __builtin_prefetch(Ar + k0 + 64, 0, 3);
    v16bf b0 = load_b_frag(Bb,               k0, half);
    v16bf b1 = load_b_frag(Bb + 16 * Kd,     k0, half);
    v16bf b2 = load_b_frag(Bb + 32 * Kd,     k0, half);
    v16bf b3 = load_b_frag(Bb + 48 * Kd,     k0, half);
    acc0 = __builtin_amdgcn_wmma_f32_16x16x32_bf16(false, a, false, b0, (short)0, acc0, false, false);
    acc1 = __builtin_amdgcn_wmma_f32_16x16x32_bf16(false, a, false, b1, (short)0, acc1, false, false);
    acc2 = __builtin_amdgcn_wmma_f32_16x16x32_bf16(false, a, false, b2, (short)0, acc2, false, false);
    acc3 = __builtin_amdgcn_wmma_f32_16x16x32_bf16(false, a, false, b3, (short)0, acc3, false, false);
  }
#pragma unroll
  for (int r = 0; r < 8; ++r) {
    int row = tm * 16 + r + half * 8;        // C/D layout: VGPR r -> M=r (+8 hi half)
    size_t idx = (size_t)row * N + tn4 * 64 + l16;
    float v0 = acc0[r], v1 = acc1[r], v2 = acc2[r], v3 = acc3[r];
    if (add) {
      v0 += add[idx]; v1 += add[idx + 16]; v2 += add[idx + 32]; v3 += add[idx + 48];
    }
    C[idx]      = v0;
    C[idx + 16] = v1;
    C[idx + 32] = v2;
    C[idx + 48] = v3;
  }
}

// ---------------- WMMA bf16 GEMM, single 16x16 tile/wave (any N%16==0) -----
__global__ __launch_bounds__(256) void gemm_bf16_wmma1(const unsigned short* __restrict__ A,
                                                       const unsigned short* __restrict__ BT,
                                                       const float* __restrict__ add,
                                                       float* __restrict__ C,
                                                       int Mr, int N, int Kd) {
  const int wave = blockIdx.x * (blockDim.x >> 5) + (threadIdx.x >> 5);
  const int tilesN = N >> 4;
  const int tiles = (Mr >> 4) * tilesN;
  if (wave >= tiles) return;
  const int tm = wave / tilesN;
  const int tn = wave - tm * tilesN;
  const int lane = threadIdx.x & 31;
  const int half = lane >> 4;
  const int l16  = lane & 15;
  const unsigned short* Ar = A  + (size_t)(tm * 16 + l16) * Kd;
  const unsigned short* Br = BT + (size_t)(tn * 16 + l16) * Kd;
  v8f acc = {};
  for (int k0 = 0; k0 < Kd; k0 += 32) {
    v16bf a = load_a_frag(Ar, k0, half);
    v16bf b = load_b_frag(Br, k0, half);
    if (k0 + 64 < Kd) __builtin_prefetch(Ar + k0 + 64, 0, 3);
    acc = __builtin_amdgcn_wmma_f32_16x16x32_bf16(false, a, false, b, (short)0, acc, false, false);
  }
#pragma unroll
  for (int r = 0; r < 8; ++r) {
    int row = tm * 16 + r + half * 8;
    size_t idx = (size_t)row * N + tn * 16 + l16;
    float v = acc[r];
    if (add) v += add[idx];
    C[idx] = v;
  }
}

// -------------------------------------- causal depthwise conv + SiLU gating
__global__ __launch_bounds__(256) void conv_gate_kernel(const float* __restrict__ up,
                                                        const float* __restrict__ gh,
                                                        const float* __restrict__ cw,
                                                        const float* __restrict__ cb,
                                                        unsigned short* __restrict__ hgb) {
  int i = blockIdx.x * blockDim.x + threadIdx.x;   // over NT*DI
  if (i >= NT * DIdim) return;
  int c = i % DIdim;
  int tok = i / DIdim;
  int t = tok % Tdim;
  float acc = cb[c];
#pragma unroll
  for (int j = 0; j < KCONV; ++j) {
    int tt = t - (KCONV - 1) + j;
    if (tt >= 0)
      acc += up[(size_t)(tok - (KCONV - 1) + j) * DIdim + c] * cw[c * KCONV + j];
  }
  float h = acc * sigmoidf_(acc);          // silu(conv + b)
  float g = gh[i];
  float o = (g * sigmoidf_(g)) * h;        // silu(gate) * h
  hgb[i] = f2bf(o);
}

// --------------------- small projections: gate / beta / decay / blend (wave per row)
__global__ __launch_bounds__(256) void small_proj_kernel(const float* __restrict__ x,
                                                         const float* __restrict__ Wg,
                                                         const float* __restrict__ Wb,
                                                         const float* __restrict__ Wa,
                                                         const float* __restrict__ A_log,
                                                         const float* __restrict__ dt_bias,
                                                         const float* __restrict__ Wbl,
                                                         float* __restrict__ gateb,
                                                         float* __restrict__ betab,
                                                         float* __restrict__ decayb,
                                                         float* __restrict__ blendb) {
  int lane = threadIdx.x & 31;
  int row = blockIdx.x * (blockDim.x >> 5) + (threadIdx.x >> 5);
  if (row >= NT) return;
  float xs[8];
#pragma unroll
  for (int j = 0; j < 8; ++j) xs[j] = x[(size_t)row * Ddim + lane + j * 32];
  auto wred = [&](float p) {
#pragma unroll
    for (int s = 16; s > 0; s >>= 1) p += __shfl_xor(p, s, 32);
    return p;
  };
  float g4[Hdim];
#pragma unroll
  for (int h = 0; h < Hdim; ++h) {
    float p = 0.f;
#pragma unroll
    for (int j = 0; j < 8; ++j) p += xs[j] * Wg[(size_t)(lane + j * 32) * Hdim + h];
    g4[h] = wred(p);
  }
  float bb[8], aa[8];
#pragma unroll
  for (int m = 0; m < Mdim; ++m)
#pragma unroll
    for (int h = 0; h < Hdim; ++h) {
      float pb = 0.f, pa = 0.f;
#pragma unroll
      for (int j = 0; j < 8; ++j) {
        int d = lane + j * 32;
        pb += xs[j] * Wb[((size_t)m * Ddim + d) * Hdim + h];
        pa += xs[j] * Wa[((size_t)m * Ddim + d) * Hdim + h];
      }
      bb[m * Hdim + h] = wred(pb);
      aa[m * Hdim + h] = wred(pa);
    }
  float bl[8];
#pragma unroll
  for (int hm = 0; hm < 8; ++hm) {
    float p = 0.f;
#pragma unroll
    for (int j = 0; j < 8; ++j) p += xs[j] * Wbl[(size_t)(lane + j * 32) * 8 + hm];
    bl[hm] = wred(p);
  }
  if (lane == 0) {
    for (int h = 0; h < Hdim; ++h) gateb[row * Hdim + h] = sigmoidf_(g4[h]);
    for (int m = 0; m < Mdim; ++m)
      for (int h = 0; h < Hdim; ++h) {
        int mh = m * Hdim + h;
        betab[row * 8 + mh] = sigmoidf_(bb[mh]);
        float alpha = aa[mh] + dt_bias[mh];
        float sp = (alpha > 20.f) ? alpha : log1pf(expf(alpha));   // softplus
        decayb[row * 8 + mh] = expf(-expf(A_log[mh]) * sp);
      }
    for (int h = 0; h < Hdim; ++h) {                                // softmax over M
      float e0 = bl[h * 2], e1 = bl[h * 2 + 1];
      float mx = fmaxf(e0, e1);
      float a = expf(e0 - mx), b = expf(e1 - mx);
      float inv = 1.f / (a + b);
      blendb[row * 8 + h * 2]     = a * inv;
      blendb[row * 8 + h * 2 + 1] = b * inv;
    }
  }
}

// ---------------------------------- rk: normalize per (b,t,h) + rotate_half
__global__ __launch_bounds__(64) void rk_kernel(const float* __restrict__ dmx,
                                                float* __restrict__ rks) {
  int row = blockIdx.x >> 2;   // / Hdim
  int h = blockIdx.x & 3;
  int d = threadIdx.x;
  __shared__ float red[64];
  __shared__ float rv[64];
  float v = dmx[(size_t)row * Ddim + h * HDdim + d];
  red[d] = v * v;
  __syncthreads();
  for (int s = 32; s > 0; s >>= 1) { if (d < s) red[d] += red[d + s]; __syncthreads(); }
  float inv = 1.0f / fmaxf(sqrtf(red[0]), 1e-12f);
  float r0 = v * inv;
  rv[d] = r0;
  __syncthreads();
  float rot = (d & 1) ? rv[d - 1] : -rv[d + 1];   // rotate_half
  size_t base = ((size_t)row * Mdim) * (Hdim * HDdim) + h * HDdim + d;
  rks[base] = r0;                   // m=0
  rks[base + Hdim * HDdim] = rot;   // m=1
}

// --------------------------------- delta-rule scan (thread = state column k)
__global__ __launch_bounds__(64) void scan_kernel(const float* __restrict__ rks,
                                                  const float* __restrict__ vals,
                                                  const float* __restrict__ decayb,
                                                  const float* __restrict__ betab,
                                                  float* __restrict__ readb) {
  int b = blockIdx.x / (Mdim * Hdim);
  int mh = blockIdx.x % (Mdim * Hdim);
  int m = mh / Hdim, h = mh % Hdim;
  int k = threadIdx.x;
  float S[HDdim];
#pragma unroll
  for (int d = 0; d < HDdim; ++d) S[d] = 0.f;
  __shared__ float wk_s[HDdim], rk_s[HDdim], v_s[HDdim];
  for (int t = 0; t < Tdim; ++t) {
    int row = b * Tdim + t;
    size_t base = ((size_t)row * Mdim + m) * (Hdim * HDdim) + h * HDdim;
    wk_s[k] = (t == 0) ? 0.f : rks[base - (size_t)Mdim * Hdim * HDdim + k];
    rk_s[k] = rks[base + k];
    v_s[k]  = vals[(size_t)row * (Mdim * Hdim * HDdim) + m * (Hdim * HDdim) + h * HDdim + k];
    __syncthreads();
    float dc = decayb[row * 8 + m * Hdim + h];
    float bt = betab[row * 8 + m * Hdim + h];
    float pred = 0.f;
#pragma unroll
    for (int d = 0; d < HDdim; ++d) pred += S[d] * wk_s[d];
    pred *= dc;                                   // S decayed before predict
    float err = (v_s[k] - pred) * bt;
    float rd = 0.f;
#pragma unroll
    for (int d = 0; d < HDdim; ++d) {
      float s = S[d] * dc + wk_s[d] * err;        // decay + rank-1 update
      S[d] = s;
      rd += s * rk_s[d];                          // read
    }
    readb[base + k] = rd;
    __syncthreads();
  }
}

// --------------------------------------- blend over M, apply gate, -> bf16 o
__global__ __launch_bounds__(256) void combine_kernel(const float* __restrict__ readb,
                                                      const float* __restrict__ blendb,
                                                      const float* __restrict__ gateb,
                                                      unsigned short* __restrict__ obf) {
  int i = blockIdx.x * blockDim.x + threadIdx.x;  // NT*D
  if (i >= NT * Ddim) return;
  int k = i & 63;
  int h = (i >> 6) & 3;
  int row = i >> 8;
  float s = 0.f;
#pragma unroll
  for (int m = 0; m < Mdim; ++m)
    s += blendb[row * 8 + h * 2 + m] *
         readb[((size_t)row * Mdim + m) * (Hdim * HDdim) + h * HDdim + k];
  s *= gateb[row * 4 + h];
  obf[i] = f2bf(s);
}

// ------------------------------------------------------------- residual add
__global__ __launch_bounds__(256) void add_kernel(float* __restrict__ x,
                                                  const float* __restrict__ t, int n) {
  int i = blockIdx.x * blockDim.x + threadIdx.x;
  if (i < n) x[i] += t[i];
}

// ===========================================================================
extern "C" void kernel_launch(void* const* d_in, const int* in_sizes, int n_in,
                              void* d_out, int out_size, void* d_ws, size_t ws_size,
                              hipStream_t stream) {
  (void)in_sizes; (void)n_in; (void)out_size; (void)ws_size;
  const int*   ids   = (const int*)d_in[0];
  const float* emb   = (const float*)d_in[1];
  const float* normw = (const float*)d_in[2];
  const float* Wup   = (const float*)d_in[3];
  const float* Wgt   = (const float*)d_in[4];
  const float* Wdn   = (const float*)d_in[5];
  const float* convw = (const float*)d_in[6];
  const float* convb = (const float*)d_in[7];
  const float* Wv    = (const float*)d_in[8];
  const float* Wg    = (const float*)d_in[9];
  const float* Wb    = (const float*)d_in[10];
  const float* Wa    = (const float*)d_in[11];
  const float* A_log = (const float*)d_in[12];
  const float* dtb   = (const float*)d_in[13];
  const float* Wbl   = (const float*)d_in[14];
  const float* Wout  = (const float*)d_in[15];
  const float* fnw   = (const float*)d_in[16];
  float* outp = (float*)d_out;

  char* wsp = (char*)d_ws;
  size_t off = 0;
  auto alloc = [&](size_t bytes) -> char* {
    char* p = wsp + off;
    off += (bytes + 255) & ~(size_t)255;
    return p;
  };
  float* xbuf  = (float*)alloc((size_t)NT * Ddim * 4);
  float* nxbuf = (float*)alloc((size_t)NT * Ddim * 4);
  unsigned short* abf = (unsigned short*)alloc((size_t)NT * DIdim * 2); // reused bf16 activations
  float* upbuf = (float*)alloc((size_t)NT * DIdim * 4);
  float* ghbuf = (float*)alloc((size_t)NT * DIdim * 4);
  float* dmx   = (float*)alloc((size_t)NT * Ddim * 4);
  float* valsb = (float*)alloc((size_t)NT * DIdim * 4);
  float* rksb  = (float*)alloc((size_t)NT * Mdim * Ddim * 4);
  float* readb = (float*)alloc((size_t)NT * Mdim * Ddim * 4);
  float* gateb = (float*)alloc((size_t)NT * Hdim * 4);
  float* betab = (float*)alloc((size_t)NT * 8 * 4);
  float* decayb= (float*)alloc((size_t)NT * 8 * 4);
  float* blendb= (float*)alloc((size_t)NT * 8 * 4);
  unsigned short* wub  = (unsigned short*)alloc((size_t)Ddim * DIdim * 2);
  unsigned short* wgb  = (unsigned short*)alloc((size_t)Ddim * DIdim * 2);
  unsigned short* wdb  = (unsigned short*)alloc((size_t)DIdim * Ddim * 2);
  unsigned short* wvb  = (unsigned short*)alloc((size_t)Ddim * DIdim * 2);
  unsigned short* wob  = (unsigned short*)alloc((size_t)Ddim * Ddim * 2);
  unsigned short* embb = (unsigned short*)alloc((size_t)VOCABd * Ddim * 2);

  const int TPB = 256;
  auto gemm = [&](const unsigned short* A, const unsigned short* BT, const float* add,
                  float* C, int Mr, int N, int Kd) {
    int tilesN = N / 16;
    if ((tilesN & 3) == 0) {
      int tiles = (Mr / 16) * (tilesN / 4);
      gemm_bf16_wmma4<<<(tiles + 7) / 8, 256, 0, stream>>>(A, BT, add, C, Mr, N, Kd);
    } else {
      int tiles = (Mr / 16) * tilesN;
      gemm_bf16_wmma1<<<(tiles + 7) / 8, 256, 0, stream>>>(A, BT, add, C, Mr, N, Kd);
    }
  };
  auto convT = [&](const float* in, unsigned short* out, int Kd, int N) {
    cvtT_bf16_kernel<<<(Kd * N + TPB - 1) / TPB, TPB, 0, stream>>>(in, out, Kd, N);
  };

  embed_kernel<<<(NT * Ddim + TPB - 1) / TPB, TPB, 0, stream>>>(ids, emb, xbuf);
  // emb (VOCAB x D) row-major is exactly the N x K transposed layout for the LM head
  cvt_bf16_kernel<<<(VOCABd * Ddim + TPB - 1) / TPB, TPB, 0, stream>>>(emb, embb, VOCABd * Ddim);

  for (int l = 0; l < Ldim; ++l) {
    convT(Wup  + (size_t)l * Ddim * DIdim, wub, Ddim, DIdim);
    convT(Wgt  + (size_t)l * Ddim * DIdim, wgb, Ddim, DIdim);
    convT(Wdn  + (size_t)l * DIdim * Ddim, wdb, DIdim, Ddim);
    convT(Wv   + (size_t)l * Ddim * (Mdim * Ddim), wvb, Ddim, Mdim * Ddim);
    convT(Wout + (size_t)l * Ddim * Ddim, wob, Ddim, Ddim);

    rmsnorm_kernel<<<NT, 256, 0, stream>>>(xbuf, normw + (size_t)l * Ddim, nxbuf, abf);
    gemm(abf, wub, nullptr, upbuf, NT, DIdim, Ddim);            // nx @ W_up
    gemm(abf, wgb, nullptr, ghbuf, NT, DIdim, Ddim);            // nx @ W_gate
    conv_gate_kernel<<<(NT * DIdim + TPB - 1) / TPB, TPB, 0, stream>>>(
        upbuf, ghbuf, convw + (size_t)l * DIdim * KCONV, convb + (size_t)l * DIdim, abf);
    gemm(abf, wdb, nullptr, dmx, NT, Ddim, DIdim);              // hg @ W_down
    cvt_bf16_kernel<<<(NT * Ddim + TPB - 1) / TPB, TPB, 0, stream>>>(dmx, abf, NT * Ddim);
    gemm(abf, wvb, nullptr, valsb, NT, Mdim * Ddim, Ddim);      // vals = dmx @ Wv
    small_proj_kernel<<<NT / 8, 256, 0, stream>>>(
        dmx, Wg + (size_t)l * Ddim * Hdim,
        Wb + (size_t)l * Mdim * Ddim * Hdim, Wa + (size_t)l * Mdim * Ddim * Hdim,
        A_log + (size_t)l * Mdim * Hdim, dtb + (size_t)l * Mdim * Hdim,
        Wbl + (size_t)l * Ddim * 8, gateb, betab, decayb, blendb);
    rk_kernel<<<NT * Hdim, 64, 0, stream>>>(dmx, rksb);
    scan_kernel<<<Bdim * Mdim * Hdim, 64, 0, stream>>>(rksb, valsb, decayb, betab, readb);
    combine_kernel<<<(NT * Ddim + TPB - 1) / TPB, TPB, 0, stream>>>(readb, blendb, gateb, abf);
    gemm(abf, wob, dmx, upbuf, NT, Ddim, Ddim);                 // o @ Wout + dmx
    add_kernel<<<(NT * Ddim + TPB - 1) / TPB, TPB, 0, stream>>>(xbuf, upbuf, NT * Ddim);
  }

  rmsnorm_kernel<<<NT, 256, 0, stream>>>(xbuf, fnw, nxbuf, abf);
  gemm(abf, embb, nullptr, outp, NT, VOCABd, Ddim);             // logits = y @ emb.T
}
